// FlowAttention_77171972374738
// MI455X (gfx1250) — compile-verified, matrix operand order checked by reading, not run on hardware
//
#include <hip/hip_runtime.h>
#include <hip/hip_bf16.h>

// ---------------------------------------------------------------------------
// FlowAttention for MI455X (gfx1250, wave32, WMMA + async global->LDS DMA).
// Pipeline:
//   cvt x,Wqkv,Wout -> bf16
//   G1: qkv = x @ Wqkv^T (+b)          [WMMA bf16 -> f32, double-buffered
//                                       global_load_async_to_lds_b128]
//   K1: qs=sig(q), ks=sig(k), sum_q/sum_k over S      (block reduce + atomics)
//   K2a: i,o per (b,s,h)   K2b: sum(q/i), sum(k/o) over S
//   K3a: i_hat,o_hat       K3b: softmax stats of o_hat over S per (b,h)
//   K4: per (b,s): M=(q/i)k^T (8x8), r = M@vw * sig(i_hat) -> values (bf16)
//   G2: out = values @ Wout^T (+b)     [same WMMA kernel]
// ---------------------------------------------------------------------------

typedef __bf16 bf16_t;
typedef __attribute__((ext_vector_type(16))) __bf16 v16bf;
typedef __attribute__((ext_vector_type(8)))  float  v8f;
typedef __attribute__((address_space(3))) unsigned int lds_uint;

union Frag { unsigned int u[8]; v16bf v; };

// ---------------------------------------------------------------------------
// f32 -> bf16 conversion (streaming)
// ---------------------------------------------------------------------------
__global__ __launch_bounds__(256) void cvt_bf16(const float* __restrict__ in,
                                                bf16_t* __restrict__ out,
                                                long n) {
  long i = (long)blockIdx.x * 256 + threadIdx.x;
  if (i < n) out[i] = (bf16_t)in[i];
}

// ---------------------------------------------------------------------------
// WMMA GEMM: C[M,N] = A[M,K](bf16) @ Bw[N,K]^T(bf16) + bias
// Block tile 128x64, BK=32, 8 waves: each wave 32x32 = 2x2 v_wmma tiles.
// Double-buffered LDS filled by GLOBAL_LOAD_ASYNC_TO_LDS_B128 (ASYNCcnt):
// the DMA of K-slab kt+1 overlaps the WMMA of slab kt; one barrier/iter.
// LDS rows padded to 40 halves (80B) so b128 ops stay 16B aligned.
// ---------------------------------------------------------------------------
__global__ __launch_bounds__(256) void gemm_bf16_wmma(
    const bf16_t* __restrict__ A, const bf16_t* __restrict__ Bw,
    const float* __restrict__ bias, float* __restrict__ C,
    int M, int N, int K) {
  __shared__ unsigned int AsU[2][128 * 20];  // 2 x (128 rows x 40 halves)
  __shared__ unsigned int BsU[2][64 * 20];   // 2 x (64 rows  x 40 halves)

  const int tid  = threadIdx.x;
  const int m0   = blockIdx.y * 128;
  const int n0   = blockIdx.x * 64;
  const int wave = tid >> 5;
  const int lane = tid & 31;
  const int mB   = (wave & 3) * 32;
  const int nB   = (wave >> 2) * 32;
  const int lrow = lane & 15;  // M (A) / N (B) within 16x16 tile
  const int lhi  = lane >> 4;  // selects K half group per ISA layout

  // Per-thread staging assignment: 3 x 16B DMA per thread per K-slab.
  const int rowA  = tid >> 2;            // A rows 0..63  (first half)
  const int rowA2 = (tid + 256) >> 2;    // A rows 64..127 (second half)
  const int cs    = (tid & 3) << 3;      // half-column segment (8 bf16)

  auto issue_async = [&](int buf, int k0) {
    const bf16_t* ga0 = A + (long)(m0 + rowA) * K + k0 + cs;
    const bf16_t* ga1 = A + (long)(m0 + rowA2) * K + k0 + cs;
    const bf16_t* gb  = Bw + (long)(n0 + rowA) * K + k0 + cs;
    unsigned la0 = (unsigned)(size_t)(lds_uint*)&AsU[buf][rowA * 20 + (cs >> 1)];
    unsigned la1 = (unsigned)(size_t)(lds_uint*)&AsU[buf][rowA2 * 20 + (cs >> 1)];
    unsigned lb  = (unsigned)(size_t)(lds_uint*)&BsU[buf][rowA * 20 + (cs >> 1)];
    asm volatile("global_load_async_to_lds_b128 %0, %1, off"
                 :: "v"(la0), "v"(ga0) : "memory");
    asm volatile("global_load_async_to_lds_b128 %0, %1, off"
                 :: "v"(la1), "v"(ga1) : "memory");
    asm volatile("global_load_async_to_lds_b128 %0, %1, off"
                 :: "v"(lb), "v"(gb) : "memory");
  };

  Frag aF[2], bF[2];
  v8f  acc[2][2] = {};

  const int nk = K >> 5;
  issue_async(0, 0);  // prime the pipe

  for (int kt = 0; kt < nk; ++kt) {
    const int cur = kt & 1;
    // Make buffer `cur` visible to all waves.
    asm volatile("s_wait_asynccnt 0x0" ::: "memory");
    __syncthreads();
    // Kick off DMA for the next K-slab into the other buffer; it lands while
    // the WMMAs below execute. (Buffer 1-cur was fully consumed before the
    // barrier above, so overwrite is safe.)
    if (kt + 1 < nk) issue_async(1 - cur, (kt + 1) << 5);

    // Build fragments per ISA 16-bit A layout (lanes 0-15: K=0..7,16..23;
    // lanes 16-31: K=8..15,24..31 — two bf16 per VGPR).
#pragma unroll
    for (int ti = 0; ti < 2; ++ti) {
      int r = mB + ti * 16 + lrow;
#pragma unroll
      for (int j = 0; j < 4; ++j) {
        aF[ti].u[j]     = AsU[cur][r * 20 + lhi * 4 + j];
        aF[ti].u[4 + j] = AsU[cur][r * 20 + 8 + lhi * 4 + j];
      }
    }
#pragma unroll
    for (int tj = 0; tj < 2; ++tj) {
      int r = nB + tj * 16 + lrow;
#pragma unroll
      for (int j = 0; j < 4; ++j) {
        bF[tj].u[j]     = BsU[cur][r * 20 + lhi * 4 + j];
        bF[tj].u[4 + j] = BsU[cur][r * 20 + 8 + lhi * 4 + j];
      }
    }
#pragma unroll
    for (int ti = 0; ti < 2; ++ti)
#pragma unroll
      for (int tj = 0; tj < 2; ++tj)
        acc[ti][tj] = __builtin_amdgcn_wmma_f32_16x16x32_bf16(
            false, aF[ti].v, false, bF[tj].v, (short)0, acc[ti][tj],
            false, false);
  }

  // C/D layout: VGPR i <-> M = i + (lane>=16 ? 8 : 0), N = lane&15.
#pragma unroll
  for (int ti = 0; ti < 2; ++ti) {
    int mrow = m0 + mB + ti * 16 + lhi * 8;
#pragma unroll
    for (int tj = 0; tj < 2; ++tj) {
      int n = n0 + nB + tj * 16 + lrow;
      float bv = bias ? bias[n] : 0.0f;
#pragma unroll
      for (int i = 0; i < 8; ++i)
        C[(long)(mrow + i) * N + n] = acc[ti][tj][i] + bv;
    }
  }
}

// ---------------------------------------------------------------------------
// K1: qs = sigmoid(q), ks = sigmoid(k); sum over S into sum_q/sum_k [B,H,D]
// grid (S/512, H, B), block 256 = 64 d-lanes x 4 s-lanes
// ---------------------------------------------------------------------------
__global__ __launch_bounds__(256) void sigmoid_sums(
    const float* __restrict__ qkv, float* __restrict__ qs,
    float* __restrict__ ks, float* __restrict__ sum_q,
    float* __restrict__ sum_k, int S) {
  int h = blockIdx.y, b = blockIdx.z;
  int s0 = blockIdx.x * 512;
  int d = threadIdx.x & 63, si = threadIdx.x >> 6;
  float aq = 0.f, ak = 0.f;
  for (int s = s0 + si; s < s0 + 512; s += 4) {
    long row = (long)b * S + s;
    float qv = qkv[row * 1536 + h * 192 + d];
    float kv = qkv[row * 1536 + h * 192 + 64 + d];
    float q1 = 1.0f / (1.0f + __expf(-qv));
    float k1 = 1.0f / (1.0f + __expf(-kv));
    qs[row * 512 + h * 64 + d] = q1;
    ks[row * 512 + h * 64 + d] = k1;
    aq += q1; ak += k1;
  }
  __shared__ float red[256];
  red[threadIdx.x] = aq; __syncthreads();
  if (si == 0)
    atomicAdd(&sum_q[(b * 8 + h) * 64 + d],
              red[d] + red[d + 64] + red[d + 128] + red[d + 192]);
  __syncthreads();
  red[threadIdx.x] = ak; __syncthreads();
  if (si == 0)
    atomicAdd(&sum_k[(b * 8 + h) * 64 + d],
              red[d] + red[d + 64] + red[d + 128] + red[d + 192]);
}

// ---------------------------------------------------------------------------
// K2a/K3a: per-(b,s,h) dot of length 64 against a per-(b,h) vector.
//   outq[row,h] = sum_d qs[row,h,d] * vq[b,h,d]
//   outk[row,h] = sum_d ks[row,h,d] * vk[b,h,d]
// grid BS, block 256 (8 waves = 8 heads, wave32 shuffle reduce)
// ---------------------------------------------------------------------------
__global__ __launch_bounds__(256) void dot_heads(
    const float* __restrict__ qs, const float* __restrict__ ks,
    const float* __restrict__ vq, const float* __restrict__ vk,
    float* __restrict__ outq, float* __restrict__ outk, int S) {
  long row = blockIdx.x;
  int b = (int)(row / S);
  int h = threadIdx.x >> 5, lane = threadIdx.x & 31;
  int d0 = lane * 2;
  long base = row * 512 + h * 64;
  int vb = (b * 8 + h) * 64;
  float a = qs[base + d0] * vq[vb + d0] + qs[base + d0 + 1] * vq[vb + d0 + 1];
  float c = ks[base + d0] * vk[vb + d0] + ks[base + d0 + 1] * vk[vb + d0 + 1];
#pragma unroll
  for (int off = 16; off > 0; off >>= 1) {
    a += __shfl_xor(a, off, 32);
    c += __shfl_xor(c, off, 32);
  }
  if (lane == 0) { outq[row * 8 + h] = a; outk[row * 8 + h] = c; }
}

// ---------------------------------------------------------------------------
// K2b: sum over S of qs/i and ks/o into sum_qi/sum_ko [B,H,D]
// ---------------------------------------------------------------------------
__global__ __launch_bounds__(256) void accum_norm2(
    const float* __restrict__ qs, const float* __restrict__ ks,
    const float* __restrict__ iA, const float* __restrict__ oA,
    float* __restrict__ sum_qi, float* __restrict__ sum_ko, int S) {
  int h = blockIdx.y, b = blockIdx.z;
  int s0 = blockIdx.x * 512;
  int d = threadIdx.x & 63, si = threadIdx.x >> 6;
  float aq = 0.f, ak = 0.f;
  for (int s = s0 + si; s < s0 + 512; s += 4) {
    long row = (long)b * S + s;
    aq += qs[row * 512 + h * 64 + d] / iA[row * 8 + h];
    ak += ks[row * 512 + h * 64 + d] / oA[row * 8 + h];
  }
  __shared__ float red[256];
  red[threadIdx.x] = aq; __syncthreads();
  if (si == 0)
    atomicAdd(&sum_qi[(b * 8 + h) * 64 + d],
              red[d] + red[d + 64] + red[d + 128] + red[d + 192]);
  __syncthreads();
  red[threadIdx.x] = ak; __syncthreads();
  if (si == 0)
    atomicAdd(&sum_ko[(b * 8 + h) * 64 + d],
              red[d] + red[d + 64] + red[d + 128] + red[d + 192]);
}

// ---------------------------------------------------------------------------
// K3b: softmax stats of o_hat over S per (b,h). grid = B*H, block 256.
// ---------------------------------------------------------------------------
__global__ __launch_bounds__(256) void softmax_stats(
    const float* __restrict__ ohat, float* __restrict__ smax,
    float* __restrict__ sden, int S) {
  int bh = blockIdx.x;
  int b = bh >> 3, h = bh & 7;
  __shared__ float red[256];
  float mx = -1e30f;
  for (int s = threadIdx.x; s < S; s += 256)
    mx = fmaxf(mx, ohat[((long)b * S + s) * 8 + h]);
  red[threadIdx.x] = mx; __syncthreads();
  for (int off = 128; off > 0; off >>= 1) {
    if (threadIdx.x < off)
      red[threadIdx.x] = fmaxf(red[threadIdx.x], red[threadIdx.x + off]);
    __syncthreads();
  }
  mx = red[0]; __syncthreads();
  float sm = 0.f;
  for (int s = threadIdx.x; s < S; s += 256)
    sm += __expf(ohat[((long)b * S + s) * 8 + h] - mx);
  red[threadIdx.x] = sm; __syncthreads();
  for (int off = 128; off > 0; off >>= 1) {
    if (threadIdx.x < off) red[threadIdx.x] += red[threadIdx.x + off];
    __syncthreads();
  }
  if (threadIdx.x == 0) { smax[bh] = mx; sden[bh] = red[0]; }
}

// ---------------------------------------------------------------------------
// K4: per (b,s): M = (q/i) ks^T  (8x8), r = (M @ vw) * sig(i_hat)
// (algebraically identical to kv = k^T vw; r = (q/i) kv, 8x fewer FLOPs)
// grid BS, block 64 (one thread per e)
// ---------------------------------------------------------------------------
__global__ __launch_bounds__(64) void per_position(
    const float* __restrict__ qs, const float* __restrict__ ks,
    const float* __restrict__ qkv, const float* __restrict__ iA,
    const float* __restrict__ ihatA, const float* __restrict__ ohatA,
    const float* __restrict__ smax, const float* __restrict__ sden,
    bf16_t* __restrict__ valsbf, int S) {
  long row = blockIdx.x;
  int b = (int)(row / S);
  int e = threadIdx.x;
  __shared__ float qi_s[8][64], ks_s[8][64], vw_s[8][64], M_s[8][8];
  __shared__ float wv[8], sg[8];
#pragma unroll
  for (int h = 0; h < 8; ++h) {
    long base = row * 512 + h * 64;
    qi_s[h][e] = qs[base + e] / iA[row * 8 + h];
    ks_s[h][e] = ks[base + e];
    vw_s[h][e] = qkv[row * 1536 + h * 192 + 128 + e];  // raw v
  }
  if (e < 8) {
    wv[e] = __expf(ohatA[row * 8 + e] - smax[b * 8 + e]) / sden[b * 8 + e];
    sg[e] = 1.0f / (1.0f + __expf(-ihatA[row * 8 + e]));
  }
  __syncthreads();
#pragma unroll
  for (int h = 0; h < 8; ++h) vw_s[h][e] *= wv[h];
  {
    int h = e >> 3, hp = e & 7;
    float m = 0.f;
    for (int d = 0; d < 64; ++d) m += qi_s[h][d] * ks_s[hp][d];
    M_s[h][hp] = m;
  }
  __syncthreads();
#pragma unroll
  for (int h = 0; h < 8; ++h) {
    float a = 0.f;
#pragma unroll
    for (int hp = 0; hp < 8; ++hp) a += M_s[h][hp] * vw_s[hp][e];
    valsbf[row * 512 + h * 64 + e] = (bf16_t)(a * sg[h]);
  }
}

// ---------------------------------------------------------------------------
extern "C" void kernel_launch(void* const* d_in, const int* in_sizes, int n_in,
                              void* d_out, int out_size, void* d_ws,
                              size_t ws_size, hipStream_t stream) {
  const float* x    = (const float*)d_in[0];
  const float* Wqkv = (const float*)d_in[1];
  const float* bqkv = (const float*)d_in[2];
  const float* Wout = (const float*)d_in[3];
  const float* bout = (const float*)d_in[4];
  float* out = (float*)d_out;

  constexpr int Bb = 4, S = 4096, E = 512, H = 8;
  constexpr int BS = Bb * S;       // 16384
  constexpr int N1 = 3 * E;        // 1536
  constexpr int K = E;             // 512

  char* ws = (char*)d_ws;
  size_t off = 0;
  auto carve = [&](size_t bytes) {
    void* p = ws + off;
    off = (off + bytes + 255) & ~(size_t)255;
    return p;
  };
  bf16_t* xbf    = (bf16_t*)carve((size_t)BS * E * 2);
  bf16_t* wqkvbf = (bf16_t*)carve((size_t)N1 * K * 2);
  bf16_t* woutbf = (bf16_t*)carve((size_t)E * K * 2);
  bf16_t* valsbf = (bf16_t*)carve((size_t)BS * E * 2);
  float*  qkv    = (float*)carve((size_t)BS * N1 * 4);
  float*  qs     = (float*)carve((size_t)BS * E * 4);
  float*  ks     = (float*)carve((size_t)BS * E * 4);
  float*  sum_q  = (float*)carve(Bb * H * 64 * 4);
  float*  sum_k  = (float*)carve(Bb * H * 64 * 4);
  float*  sum_ko = (float*)carve(Bb * H * 64 * 4);
  float*  sum_qi = (float*)carve(Bb * H * 64 * 4);
  float*  iA     = (float*)carve((size_t)BS * H * 4);
  float*  oA     = (float*)carve((size_t)BS * H * 4);
  float*  ihatA  = (float*)carve((size_t)BS * H * 4);
  float*  ohatA  = (float*)carve((size_t)BS * H * 4);
  float*  smax   = (float*)carve(Bb * H * 4);
  float*  sden   = (float*)carve(Bb * H * 4);

  // zero the atomic accumulators (every call: graph replays must be identical)
  hipMemsetAsync(sum_q, 0, 4 * (size_t)(Bb * H * 64) * 4 + 512, stream);

  // conversions
  cvt_bf16<<<(BS * (long)E + 255) / 256, 256, 0, stream>>>(x, xbf, (long)BS * E);
  cvt_bf16<<<((long)N1 * K + 255) / 256, 256, 0, stream>>>(Wqkv, wqkvbf, (long)N1 * K);
  cvt_bf16<<<((long)E * K + 255) / 256, 256, 0, stream>>>(Wout, woutbf, (long)E * K);

  // G1: qkv = x @ Wqkv^T + b_qkv
  gemm_bf16_wmma<<<dim3(N1 / 64, BS / 128), 256, 0, stream>>>(
      xbf, wqkvbf, bqkv, qkv, BS, N1, K);

  // K1
  sigmoid_sums<<<dim3(S / 512, H, Bb), 256, 0, stream>>>(qkv, qs, ks, sum_q,
                                                         sum_k, S);
  // K2a: i = <qs, sum_k>, o = <ks, sum_q>
  dot_heads<<<BS, 256, 0, stream>>>(qs, ks, sum_k, sum_q, iA, oA, S);
  // K2b: sum_qi = sum_s qs/i, sum_ko = sum_s ks/o
  accum_norm2<<<dim3(S / 512, H, Bb), 256, 0, stream>>>(qs, ks, iA, oA,
                                                        sum_qi, sum_ko, S);
  // K3a: i_hat = <qs, sum_ko>, o_hat = <ks, sum_qi>
  dot_heads<<<BS, 256, 0, stream>>>(qs, ks, sum_ko, sum_qi, ihatA, ohatA, S);
  // K3b
  softmax_stats<<<Bb * H, 256, 0, stream>>>(ohatA, smax, sden, S);
  // K4
  per_position<<<BS, 64, 0, stream>>>(qs, ks, qkv, iA, ihatA, ohatA, smax,
                                      sden, valsbf, S);
  // G2: out = values @ Wout^T + b_out
  gemm_bf16_wmma<<<dim3(E / 64, BS / 128), 256, 0, stream>>>(
      valsbf, woutbf, bout, out, BS, E, K);
}
